// GatedCrossAttentionFuse_82197084111488
// MI455X (gfx1250) — matrix operand order repaired and checked by reading, not compiled, not run on hardware
//
#include <hip/hip_runtime.h>
#include <hip/hip_bf16.h>

typedef __attribute__((ext_vector_type(16))) _Float16 v16h;
typedef __attribute__((ext_vector_type(8)))  float    v8f;

#define B_    4
#define C_    128
#define H_    180
#define W_    180
#define HW_   (H_ * W_)
#define CCAM_ 256
#define HEADS_ 4
#define DH_   32
#define HD_   (HEADS_ * DH_)
#define N_    20000
#define GAMMA_ 0.08f
#define EPS_   1e-6f
#define SCALE_ 0.17677669529663687f  /* 1/sqrt(32) */

// ---------------- workspace layout (bytes) ----------------
#define OFF_DELTA 0ull
#define SZ_DELTA  ((size_t)B_ * C_ * HW_ * 4)          // 66,355,200
#define OFF_HITS  (OFF_DELTA + SZ_DELTA)
#define SZ_HITS   ((size_t)B_ * HW_ * 4)               // 518,400
#define OFF_MEAN  (OFF_HITS + SZ_HITS)
#define SZ_MEAN   ((size_t)B_ * C_ * 4)                // 2,048
#define OFF_CNT   (OFF_MEAN + SZ_MEAN)
#define SZ_CNT    256ull
#define OFF_LIN   (OFF_CNT + SZ_CNT)                   // zero-init region ends here
#define SZ_LIN    ((size_t)B_ * N_ * 4)
#define OFF_QG    (OFF_LIN + SZ_LIN)
#define SZ_QG     ((size_t)B_ * HD_ * N_ * 4)          // 40,960,000
#define OFF_KK    (OFF_QG + SZ_QG)
#define OFF_VV    (OFF_KK + SZ_QG)
#define OFF_ATTN  (OFF_VV + SZ_QG)
#define SZ_ATTN   ((size_t)B_ * HEADS_ * N_ * 4)
#define WS_NEEDED (OFF_ATTN + SZ_ATTN)

// Non-returning fp32 global atomic add (CDNA5 VGLOBAL encoding).
__device__ __forceinline__ void gatomic_add_f32(float* p, float v) {
  asm volatile("global_atomic_add_f32 %0, %1, off" : : "v"(p), "v"(v) : "memory");
}

// A-matrix (16x32 f16) per-lane K index for v16h element e, lane-group g.
__device__ __forceinline__ int a_kmap(int e, int g) {
  int p = e >> 1;
  return ((p & 3) << 1) + (e & 1) + (g << 3) + ((p >> 2) << 4);
}

// ---------------------------------------------------------------------------
__global__ void k_zero(float* p, size_t n) {
  size_t i = (size_t)blockIdx.x * blockDim.x + threadIdx.x;
  if (i < n) p[i] = 0.0f;
}

__global__ void k_prep(const int* __restrict__ cidx, int* __restrict__ lin,
                       unsigned* __restrict__ hits) {
  int idx = blockIdx.x * blockDim.x + threadIdx.x;
  if (idx >= B_ * N_) return;
  int b = idx / N_;
  int i = cidx[2 * idx + 0];
  int j = cidx[2 * idx + 1];
  i = min(max(i, 0), H_ - 1);
  j = min(max(j, 0), W_ - 1);
  int l = i * W_ + j;
  lin[idx] = l;
  atomicAdd(&hits[b * HW_ + l], 1u);
}

// q_g[b,m,n] = q_w[m,:] . lidar[b,:,pix(n)] + q_bias[m]
__global__ void k_gemm_q(const float* __restrict__ lidar, const float* __restrict__ q_w,
                         const float* __restrict__ q_bias, const int* __restrict__ lin,
                         float* __restrict__ qg) {
  int wid  = blockIdx.x * (blockDim.x >> 5) + (threadIdx.x >> 5);
  int lane = threadIdx.x & 31;
  int g = lane >> 4, lr = lane & 15;
  int b   = wid / (8 * 1250);
  int rem = wid % (8 * 1250);
  int m0 = (rem / 1250) * 16;
  int n0 = (rem % 1250) * 16;
  int n  = n0 + lr;
  int pix = lin[b * N_ + n];
  const float* Xb = lidar + (size_t)b * C_ * HW_;
  const float* Ar = q_w + (size_t)(m0 + lr) * C_;
  v8f acc = {};
  for (int kt = 0; kt < C_; kt += 32) {
    v16h A, Bm;
#pragma unroll
    for (int e = 0; e < 16; ++e) {
      int ka = kt + a_kmap(e, g);
      int kb = kt + a_kmap(e, g);
      A[e]  = (_Float16)Ar[ka];
      Bm[e] = (_Float16)Xb[(size_t)kb * HW_ + pix];
    }
    acc = __builtin_amdgcn_wmma_f32_16x16x32_f16(false, A, false, Bm, (short)0, acc,
                                                 false, false);
  }
#pragma unroll
  for (int r = 0; r < 8; ++r) {
    int m = m0 + r + (g << 3);
    qg[((size_t)b * HD_ + m) * N_ + n] = acc[r] + q_bias[m];
  }
}

// kk/vv[b,m,n] = {k_w,v_w}[m,:] . cam[b,n,:]
__global__ void k_gemm_kv(const float* __restrict__ cam, const float* __restrict__ k_w,
                          const float* __restrict__ v_w, float* __restrict__ kk,
                          float* __restrict__ vv) {
  int wid  = blockIdx.x * (blockDim.x >> 5) + (threadIdx.x >> 5);
  int lane = threadIdx.x & 31;
  int g = lane >> 4, lr = lane & 15;
  int b   = wid / (8 * 1250);
  int rem = wid % (8 * 1250);
  int m0 = (rem / 1250) * 16;
  int n0 = (rem % 1250) * 16;
  int n  = n0 + lr;
  const float* camrow = cam + ((size_t)b * N_ + n) * CCAM_;
  const float* Akr = k_w + (size_t)(m0 + lr) * CCAM_;
  const float* Avr = v_w + (size_t)(m0 + lr) * CCAM_;
  v8f ak = {}, av = {};
  for (int kt = 0; kt < CCAM_; kt += 32) {
    v16h Ak, Av, Bm;
#pragma unroll
    for (int e = 0; e < 16; ++e) {
      int ka = kt + a_kmap(e, g);
      int kb = kt + a_kmap(e, g);
      Ak[e] = (_Float16)Akr[ka];
      Av[e] = (_Float16)Avr[ka];
      Bm[e] = (_Float16)camrow[kb];
    }
    ak = __builtin_amdgcn_wmma_f32_16x16x32_f16(false, Ak, false, Bm, (short)0, ak,
                                                false, false);
    av = __builtin_amdgcn_wmma_f32_16x16x32_f16(false, Av, false, Bm, (short)0, av,
                                                false, false);
  }
#pragma unroll
  for (int r = 0; r < 8; ++r) {
    int m = m0 + r + (g << 3);
    kk[((size_t)b * HD_ + m) * N_ + n] = ak[r];
    vv[((size_t)b * HD_ + m) * N_ + n] = av[r];
  }
}

__global__ void k_logits(const float* __restrict__ qg, const float* __restrict__ kk,
                         float* __restrict__ attn) {
  int idx = blockIdx.x * blockDim.x + threadIdx.x;
  if (idx >= B_ * HEADS_ * N_) return;
  int n = idx % N_;
  int bh = idx / N_;
  int b = bh / HEADS_, h = bh % HEADS_;
  const float* qp = qg + ((size_t)(b * HD_ + h * DH_)) * N_ + n;
  const float* kp = kk + ((size_t)(b * HD_ + h * DH_)) * N_ + n;
  float s = 0.0f;
#pragma unroll
  for (int d = 0; d < DH_; ++d) s += qp[(size_t)d * N_] * kp[(size_t)d * N_];
  attn[idx] = s * SCALE_;
}

__global__ void k_softmax(float* __restrict__ attn, const float* __restrict__ gate) {
  __shared__ float red[256];
  int bh = blockIdx.x;
  int b = bh / HEADS_;
  float* row = attn + (size_t)bh * N_;
  const float* grow = gate + (size_t)b * N_;
  int t = threadIdx.x;
  float mx = -1e30f;
  for (int n = t; n < N_; n += 256) mx = fmaxf(mx, row[n]);
  red[t] = mx; __syncthreads();
  for (int s = 128; s > 0; s >>= 1) { if (t < s) red[t] = fmaxf(red[t], red[t + s]); __syncthreads(); }
  mx = red[0]; __syncthreads();
  float sum = 0.0f;
  for (int n = t; n < N_; n += 256) sum += expf(row[n] - mx);
  red[t] = sum; __syncthreads();
  for (int s = 128; s > 0; s >>= 1) { if (t < s) red[t] += red[t + s]; __syncthreads(); }
  float inv = 1.0f / red[0]; __syncthreads();
  for (int n = t; n < N_; n += 256) row[n] = expf(row[n] - mx) * inv * grow[n];
}

// out_tok[b,c,n] = out_w[c,:] . (attn[b,o/32,n]*vv[b,o,n]); scatter-add into delta
__global__ void k_gemm_out(const float* __restrict__ out_w, const float* __restrict__ attn,
                           const float* __restrict__ vv, const int* __restrict__ lin,
                           float* __restrict__ delta) {
  int wid  = blockIdx.x * (blockDim.x >> 5) + (threadIdx.x >> 5);
  int lane = threadIdx.x & 31;
  int g = lane >> 4, lr = lane & 15;
  int b   = wid / (8 * 1250);
  int rem = wid % (8 * 1250);
  int m0 = (rem / 1250) * 16;
  int n0 = (rem % 1250) * 16;
  int n  = n0 + lr;
  int pix = lin[b * N_ + n];
  const float* Ar = out_w + (size_t)(m0 + lr) * HD_;
  v8f acc = {};
  for (int kt = 0; kt < HD_; kt += 32) {
    int head = kt >> 5;  // kb in [kt, kt+32) -> single head per step
    float aval = attn[((size_t)b * HEADS_ + head) * N_ + n];
    v16h A, Bm;
#pragma unroll
    for (int e = 0; e < 16; ++e) {
      int ka = kt + a_kmap(e, g);
      int kb = kt + a_kmap(e, g);
      A[e]  = (_Float16)Ar[ka];
      Bm[e] = (_Float16)(aval * vv[((size_t)b * HD_ + kb) * N_ + n]);
    }
    acc = __builtin_amdgcn_wmma_f32_16x16x32_f16(false, A, false, Bm, (short)0, acc,
                                                 false, false);
  }
#pragma unroll
  for (int r = 0; r < 8; ++r) {
    int m = m0 + r + (g << 3);
    gatomic_add_f32(&delta[((size_t)b * C_ + m) * HW_ + pix], acc[r]);
  }
}

__global__ void k_count(const unsigned* __restrict__ hits, float* __restrict__ cnt) {
  __shared__ int red[256];
  int b = blockIdx.x, t = threadIdx.x;
  int c = 0;
  for (int i = t; i < HW_; i += 256) c += (hits[b * HW_ + i] != 0u) ? 1 : 0;
  red[t] = c; __syncthreads();
  for (int s = 128; s > 0; s >>= 1) { if (t < s) red[t] += red[t + s]; __syncthreads(); }
  if (t == 0) cnt[b] = (float)red[0];
}

__global__ void k_mean(const float* __restrict__ delta, const float* __restrict__ cnt,
                       float* __restrict__ mean) {
  __shared__ float red[256];
  int bc = blockIdx.x, t = threadIdx.x;
  int b = bc / C_;
  const float* row = delta + (size_t)bc * HW_;
  float s = 0.0f;
  for (int i = t; i < HW_; i += 256) s += row[i];
  red[t] = s; __syncthreads();
  for (int st = 128; st > 0; st >>= 1) { if (t < st) red[t] += red[t + st]; __syncthreads(); }
  if (t == 0) mean[bc] = red[0] / (cnt[b] + EPS_);
}

__global__ void k_final(const float* __restrict__ lidar, const float* __restrict__ delta,
                        const unsigned* __restrict__ hits, const float* __restrict__ mean,
                        const float* __restrict__ ralpha, float* __restrict__ out) {
  size_t idx = (size_t)blockIdx.x * blockDim.x + threadIdx.x;
  if (idx >= (size_t)B_ * C_ * HW_) return;
  int hw = (int)(idx % HW_);
  int bc = (int)(idx / HW_);
  int b = bc / C_;
  float m = (hits[b * HW_ + hw] != 0u) ? mean[bc] : 0.0f;
  out[idx] = lidar[idx] + (delta[idx] - m) * ralpha[hw] * GAMMA_;
}

// ---------------------------------------------------------------------------
extern "C" void kernel_launch(void* const* d_in, const int* in_sizes, int n_in,
                              void* d_out, int out_size, void* d_ws, size_t ws_size,
                              hipStream_t stream) {
  (void)in_sizes; (void)n_in; (void)out_size;
  if (ws_size < WS_NEEDED) return;  // insufficient scratch: bail safely

  const float* lidar  = (const float*)d_in[0];
  const float* cam    = (const float*)d_in[1];
  const float* gate   = (const float*)d_in[2];
  const float* ralpha = (const float*)d_in[3];
  const float* q_w    = (const float*)d_in[4];
  const float* q_b    = (const float*)d_in[5];
  const float* k_w    = (const float*)d_in[6];
  const float* v_w    = (const float*)d_in[7];
  const float* out_w  = (const float*)d_in[8];
  const int*   cidx   = (const int*)d_in[9];
  float* out = (float*)d_out;

  char* ws = (char*)d_ws;
  float*    delta = (float*)(ws + OFF_DELTA);
  unsigned* hits  = (unsigned*)(ws + OFF_HITS);
  float*    mean  = (float*)(ws + OFF_MEAN);
  float*    cnt   = (float*)(ws + OFF_CNT);
  int*      lin   = (int*)(ws + OFF_LIN);
  float*    qg    = (float*)(ws + OFF_QG);
  float*    kk    = (float*)(ws + OFF_KK);
  float*    vv    = (float*)(ws + OFF_VV);
  float*    attn  = (float*)(ws + OFF_ATTN);

  // 1) zero delta/hits/mean/cnt (one contiguous prefix)
  size_t zwords = OFF_LIN / 4;
  k_zero<<<(unsigned)((zwords + 255) / 256), 256, 0, stream>>>((float*)ws, zwords);

  // 2) indices + hit counts
  k_prep<<<(B_ * N_ + 255) / 256, 256, 0, stream>>>(cidx, lin, hits);

  // 3) q projection at gathered pixels (WMMA)
  const int gemm_waves  = B_ * (HD_ / 16) * (N_ / 16);  // 40000 -> exact
  const int gemm_blocks = gemm_waves / 4;               // 128 threads = 4 waves
  k_gemm_q<<<gemm_blocks, 128, 0, stream>>>(lidar, q_w, q_b, lin, qg);

  // 4) k and v projections (WMMA, shared B operand)
  k_gemm_kv<<<gemm_blocks, 128, 0, stream>>>(cam, k_w, v_w, kk, vv);

  // 5) attention logits + softmax over N + gate
  k_logits<<<(B_ * HEADS_ * N_ + 255) / 256, 256, 0, stream>>>(qg, kk, attn);
  k_softmax<<<B_ * HEADS_, 256, 0, stream>>>(attn, gate);

  // 6) output projection (WMMA) with fused attn*v, scatter-add via fp32 atomics
  k_gemm_out<<<gemm_blocks, 128, 0, stream>>>(out_w, attn, vv, lin, delta);

  // 7) hit counts per batch, per-(b,c) mean, final blend
  k_count<<<B_, 256, 0, stream>>>(hits, cnt);
  k_mean<<<B_ * C_, 256, 0, stream>>>(delta, cnt, mean);
  k_final<<<(unsigned)(((size_t)B_ * C_ * HW_ + 255) / 256), 256, 0, stream>>>(
      lidar, delta, hits, mean, ralpha, out);
}
